// AxialTemporalRKAInterleavedLayer_22247930593751
// MI455X (gfx1250) — matrix operand order repaired and checked by reading, not compiled, not run on hardware
//
#include <hip/hip_runtime.h>
#include <math.h>

// ---------------- CDNA5 WMMA types ----------------
typedef __bf16 bf16;
typedef bf16 v16bf __attribute__((ext_vector_type(16)));
typedef float v8f  __attribute__((ext_vector_type(8)));
typedef unsigned int u32x4 __attribute__((ext_vector_type(4)));
typedef unsigned int v4u  __attribute__((ext_vector_type(4)));
typedef int          v8i  __attribute__((ext_vector_type(8)));
typedef int          v4i  __attribute__((ext_vector_type(4)));
union Frag16 { v16bf v; u32x4 u[2]; };

#if __has_builtin(__builtin_amdgcn_tensor_load_to_lds)
#define HAS_TDM 1
#else
#define HAS_TDM 0
#endif

// ---------------- problem dims (hardcoded from reference) ----------------
constexpr int       kB    = 2;
constexpr int       kT    = 8;
constexpr int       kH    = 32;
constexpr int       kW    = 32;
constexpr int       kC    = 384;
constexpr int       kNH   = 8;
constexpr int       kDFF  = 1536;
constexpr int       kHD   = 48;     // head dim
constexpr int       kHDP  = 64;     // head dim padded to K%32==0
constexpr int       kL    = 256;    // sequence length per axial pass (T*W == T*H)
constexpr int       kNB   = 512;    // head-batches = (B*H)*NH = (B*W)*NH
constexpr int       kQKV  = 1152;   // 3*C
constexpr long long kMTOK = 16384;  // tokens per path = B*H*L = B*W*L

__device__ __forceinline__ float gelu_exact(float x) {
  return 0.5f * x * (1.0f + erff(x * 0.70710678118654752440f));
}
__device__ __forceinline__ float sigmoidf_(float x) {
  return 1.0f / (1.0f + expf(-x));
}

// =====================================================================
// TDM helper: async-load a [tileRows x tileK] bf16 tile (row stride
// strideElems) from global into LDS at byte offset ldsOff.
// D# packing per CDNA5 ISA ch.8 (group0 128b, group1 256b, 2-D tensor).
// =====================================================================
#if HAS_TDM
__device__ __forceinline__ void tdm_load_tile(unsigned int ldsOff, const bf16* g,
                                              unsigned int tensorK, unsigned int tensorRows,
                                              unsigned int tileK, unsigned int tileRows,
                                              unsigned int strideElems) {
  unsigned long long ga = (unsigned long long)(uintptr_t)g;
  v4u g0;
  g0[0] = 1u;                                            // count=1 (valid user desc)
  g0[1] = ldsOff;                                        // LDS byte address
  g0[2] = (unsigned int)(ga & 0xFFFFFFFFu);              // global addr [31:0]
  g0[3] = (unsigned int)((ga >> 32) & 0x01FFFFFFu) | (2u << 30);  // addr[56:32] | type=2
  v8i g1;
  g1[0] = (int)(1u << 16);                               // data_size=1 -> 2 bytes/elem
  g1[1] = (int)((tensorK & 0xFFFFu) << 16);              // tensor_dim0[15:0]
  g1[2] = (int)((tensorK >> 16) | ((tensorRows & 0xFFFFu) << 16));
  g1[3] = (int)((tensorRows >> 16) | (tileK << 16));     // tile_dim0
  g1[4] = (int)(tileRows & 0xFFFFu);                     // tile_dim1 (tile_dim2=0)
  g1[5] = (int)strideElems;                              // tensor_dim0_stride[31:0]
  g1[6] = 0;
  g1[7] = 0;
  v4i gz = {};
#if defined(__clang_major__) && __clang_major__ >= 23
  v8i gz8 = {};
  __builtin_amdgcn_tensor_load_to_lds(g0, g1, gz, gz, gz8, 0);
#else
  __builtin_amdgcn_tensor_load_to_lds(g0, g1, gz, gz, 0);
#endif
}
#endif

// =====================================================================
// LDS-staged, double-buffered WMMA GEMM:
//   Cout[M,N] = A[M,K](bf16) @ Wt[N,K]^T (bf16) + bias, optional GELU.
// Block = 128 threads (4 waves) -> 64(M) x 64(N) tile, K-step 32.
// Tiles moved global->LDS by the Tensor Data Mover (wave 0 issues,
// TENSORcnt-fenced), compute overlaps the next tile's DMA.
// All five LDS fragments are fetched before the WMMA cluster so the
// four independent-accumulator WMMAs issue behind a single DScnt wait.
// =====================================================================
template <int ACT>
__global__ void k_gemm_tdm(const bf16* __restrict__ A, const bf16* __restrict__ Wt,
                           const float* __restrict__ bias, float* __restrict__ Cout,
                           int M, int N, int K) {
  __shared__ bf16 sA[2][64 * 32];
  __shared__ bf16 sB[2][64 * 32];

  const int tid  = threadIdx.x;
  const int lane = tid & 31;
  const int wave = tid >> 5;
  const int n0 = blockIdx.x * 64;
  const int m0 = blockIdx.y * 64;

  const int  row16 = lane & 15;
  const bool hi    = (lane >= 16);
  const int  akb   = hi ? 8 : 0;    // A 16-bit 16x32 layout (ISA 7.12.2)
  const int  bkb   = hi ? 8 : 0;    // B 32x16: same K layout as A

  const int KT = K / 32;

  auto issue_load = [&](int buf, int kt) {
#if HAS_TDM
    if (wave == 0) {
      tdm_load_tile((unsigned int)(uintptr_t)&sA[buf][0],
                    A + (long long)m0 * K + kt * 32,
                    (unsigned)K, (unsigned)M, 32u, 64u, (unsigned)K);
      tdm_load_tile((unsigned int)(uintptr_t)&sB[buf][0],
                    Wt + (long long)n0 * K + kt * 32,
                    (unsigned)K, (unsigned)N, 32u, 64u, (unsigned)K);
    }
#else
    // cooperative fallback: 128 threads x half-row (16 bf16 = 32B) per tile
    {
      int r = tid >> 1, half = tid & 1;
      const bf16* srcA = A + (long long)(m0 + r) * K + kt * 32 + half * 16;
      const bf16* srcB = Wt + (long long)(n0 + r) * K + kt * 32 + half * 16;
      bf16* dstA = &sA[buf][r * 32 + half * 16];
      bf16* dstB = &sB[buf][r * 32 + half * 16];
      *(u32x4*)(dstA)     = *(const u32x4*)(srcA);
      *(u32x4*)(dstA + 8) = *(const u32x4*)(srcA + 8);
      *(u32x4*)(dstB)     = *(const u32x4*)(srcB);
      *(u32x4*)(dstB + 8) = *(const u32x4*)(srcB + 8);
    }
#endif
  };
  auto fence_load = [&]() {
#if HAS_TDM
    if (wave == 0) __builtin_amdgcn_s_wait_tensorcnt(0);
#endif
    __syncthreads();
  };

  v8f acc[4] = {};
  issue_load(0, 0);
  fence_load();

  for (int kt = 0; kt < KT; ++kt) {
    const int p = kt & 1;
    if (kt + 1 < KT) issue_load(1 - p, kt + 1);

    // fetch all fragments first (one DScnt wait), then fire 4 WMMAs
    Frag16 a, b[4];
    const bf16* aP = &sA[p][(wave * 16 + row16) * 32 + akb];
    a.u[0] = *(const u32x4*)(aP);        // K = akb   .. akb+7
    a.u[1] = *(const u32x4*)(aP + 16);   // K = akb+16.. akb+23
#pragma unroll
    for (int t = 0; t < 4; ++t) {
      const bf16* bP = &sB[p][(t * 16 + row16) * 32 + bkb];
      b[t].u[0] = *(const u32x4*)(bP);
      b[t].u[1] = *(const u32x4*)(bP + 16);
    }
#pragma unroll
    for (int t = 0; t < 4; ++t) {
      acc[t] = __builtin_amdgcn_wmma_f32_16x16x32_bf16(
          false, a.v, false, b[t].v, (short)0, acc[t], false, false);
    }
    if (kt + 1 < KT) fence_load();
  }

#pragma unroll
  for (int t = 0; t < 4; ++t) {
    const int n  = n0 + t * 16 + row16;
    const float bv = bias ? bias[n] : 0.0f;
#pragma unroll
    for (int r = 0; r < 8; ++r) {
      const int m = m0 + wave * 16 + r + (hi ? 8 : 0);
      float v = acc[t][r] + bv;
      if (ACT == 1) v = gelu_exact(v);
      Cout[(long long)m * N + n] = v;
    }
  }
}

// =====================================================================
// Direct-from-global WMMA GEMM (used for small-N GEMMs, N multiple of 16):
//   Cout = A[M,K] @ Wt[N,K]^T + bias ; batched via grid.z strides.
// =====================================================================
template <int NT, int ACT>
__global__ void k_gemm_bf16(const bf16* __restrict__ A, const bf16* __restrict__ Wt,
                            const float* __restrict__ bias, float* __restrict__ Cout,
                            int M, int N, int K,
                            long long sA, long long sW, long long sC) {
  const int lane = threadIdx.x & 31;
  const int wave = threadIdx.x >> 5;
  const int n0   = blockIdx.x * (16 * NT);
  const int m0   = blockIdx.y * 64 + wave * 16;
  const long long batch = blockIdx.z;
  A    += batch * sA;
  Wt   += batch * sW;
  Cout += batch * sC;

  const int  row = lane & 15;
  const bool hi  = (lane >= 16);
  const int  akb = hi ? 8 : 0;
  const int  bkb = hi ? 8 : 0;

  v8f acc[NT] = {};
  const bf16* aRow = A + (long long)(m0 + row) * K;
  for (int k0 = 0; k0 < K; k0 += 32) {
    Frag16 a;
    a.u[0] = *(const u32x4*)(aRow + k0 + akb);
    a.u[1] = *(const u32x4*)(aRow + k0 + akb + 16);
#pragma unroll
    for (int t = 0; t < NT; ++t) {
      const bf16* wRow = Wt + (long long)(n0 + t * 16 + row) * K + k0 + bkb;
      Frag16 b;
      b.u[0] = *(const u32x4*)(wRow);
      b.u[1] = *(const u32x4*)(wRow + 16);
      acc[t] = __builtin_amdgcn_wmma_f32_16x16x32_bf16(
          false, a.v, false, b.v, (short)0, acc[t], false, false);
    }
  }
#pragma unroll
  for (int t = 0; t < NT; ++t) {
    const int n  = n0 + t * 16 + row;
    const float bv = bias ? bias[n] : 0.0f;
#pragma unroll
    for (int r = 0; r < 8; ++r) {
      const int m = m0 + r + (hi ? 8 : 0);
      float v = acc[t][r] + bv;
      if (ACT == 1) v = gelu_exact(v);
      Cout[(long long)m * N + n] = v;
    }
  }
}

// ---------------- elementwise / layout kernels ----------------
__global__ void k_cvt_bf16(const float* __restrict__ in, bf16* __restrict__ out, long long n) {
  long long id = (long long)blockIdx.x * blockDim.x + threadIdx.x;
  if (id < n) out[id] = (bf16)in[id];
}

__global__ void k_cvt_pad(const float* __restrict__ in, bf16* __restrict__ out,
                          int R, int K, int Kp) {
  long long id = (long long)blockIdx.x * blockDim.x + threadIdx.x;
  if (id >= (long long)R * Kp) return;
  int c = (int)(id % Kp);
  int r = (int)(id / Kp);
  out[id] = (c < K) ? (bf16)in[(long long)r * K + c] : (bf16)0.0f;
}

// tw[(b*H+h)*L + (t*W+w)][c] = x[b][t*H*W + h*W + w][c]
__global__ void k_make_tw(const float* __restrict__ x, float* __restrict__ out) {
  long long id = (long long)blockIdx.x * blockDim.x + threadIdx.x;
  if (id >= kMTOK * kC) return;
  int c = (int)(id % kC);
  long long r = id / kC;
  int l  = (int)(r % kL);
  int nh = (int)(r / kL);
  int t = l / kW, w = l % kW;
  int b = nh / kH, h = nh % kH;
  long long src = (((long long)b * (kT * kH * kW)) + (long long)t * kH * kW + h * kW + w) * kC + c;
  out[id] = x[src];
}

// th[(b*W+w)*L + (t*H+h)][c] = ntw[(b*H+h)*L + (t*W+w)][c]
__global__ void k_make_th(const float* __restrict__ ntw, float* __restrict__ out) {
  long long id = (long long)blockIdx.x * blockDim.x + threadIdx.x;
  if (id >= kMTOK * kC) return;
  int c = (int)(id % kC);
  long long r = id / kC;
  int l  = (int)(r % kL);
  int nw = (int)(r / kL);
  int t = l / kH, h = l % kH;
  int b = nw / kW, w = nw % kW;
  long long src = (((long long)(b * kH + h)) * kL + (t * kW + w)) * kC + c;
  out[id] = ntw[src];
}

// final[b][t*H*W + h*W + w][c] = nth[(b*W+w)*L + (t*H+h)][c]
__global__ void k_make_final(const float* __restrict__ nth, float* __restrict__ out) {
  long long id = (long long)blockIdx.x * blockDim.x + threadIdx.x;
  if (id >= kMTOK * kC) return;
  int c = (int)(id % kC);
  long long r = id / kC;
  int pos = (int)(r % (kT * kH * kW));
  int b   = (int)(r / (kT * kH * kW));
  int t  = pos / (kH * kW);
  int hw = pos % (kH * kW);
  int h = hw / kW, w = hw % kW;
  long long src = (((long long)(b * kW + w)) * kL + (t * kH + h)) * kC + c;
  out[id] = nth[src];
}

// qkv f32 -> qh_pad bf16 [NB*L, HDP], kh bf16 [NB,L,HD], vT bf16 [NB,HD,L]
__global__ void k_split_heads(const float* __restrict__ qkv, bf16* __restrict__ qh,
                              bf16* __restrict__ kh, bf16* __restrict__ vT) {
  long long id = (long long)blockIdx.x * blockDim.x + threadIdx.x;
  if (id >= (long long)kNB * kL * kHDP) return;
  int d = (int)(id % kHDP);
  long long r = id / kHDP;
  int l  = (int)(r % kL);
  int nb = (int)(r / kL);
  int n = nb / kNH, hh = nb % kNH;
  bf16* qdst = qh + ((long long)nb * kL + l) * kHDP;
  if (d < kHD) {
    const float* base = qkv + ((long long)n * kL + l) * kQKV + hh * kHD;
    qdst[d] = (bf16)base[d];
    kh[((long long)nb * kL + l) * kHD + d] = (bf16)base[kC + d];
    vT[((long long)nb * kHD + d) * kL + l] = (bf16)base[2 * kC + d];
  } else {
    qdst[d] = (bf16)0.0f;
  }
}

// per row of 48: exact GELU then layernorm with qln_w/qln_b (in-place)
__global__ void k_qt_post(float* __restrict__ qt, const float* __restrict__ w,
                          const float* __restrict__ b) {
  long long row = blockIdx.x;
  int t = threadIdx.x;                       // 64 threads
  float* r = qt + row * kHD;
  bool act = (t < kHD);
  float v = 0.0f;
  if (act) v = gelu_exact(r[t]);
  __shared__ float s1[64], s2[64];
  s1[t] = act ? v : 0.0f;
  s2[t] = act ? v * v : 0.0f;
  __syncthreads();
  for (int s = 32; s > 0; s >>= 1) {
    if (t < s) { s1[t] += s1[t + s]; s2[t] += s2[t + s]; }
    __syncthreads();
  }
  float mean = s1[0] / (float)kHD;
  float var  = s2[0] / (float)kHD - mean * mean;
  float inv  = rsqrtf(var + 1e-5f);
  if (act) r[t] = (v - mean) * inv * w[t] + b[t];
}

// q_rel[nb][d] = scaling * mean_l qt[nb,l,d]
__global__ void k_qrel(const float* __restrict__ qt, float* __restrict__ qrel) {
  int id = blockIdx.x * blockDim.x + threadIdx.x;
  if (id >= kNB * kHD) return;
  int nb = id / kHD, d = id % kHD;
  const float* p = qt + ((long long)nb * kL) * kHD + d;
  float s = 0.0f;
  for (int l = 0; l < kL; ++l) s += p[(long long)l * kHD];
  qrel[id] = s * (1.0f / (float)kL) * 0.14433756729740643f;  // 48^-0.5
}

// Wk[nb][l] = dot(q_rel[nb], kh[nb,l,:])
__global__ void k_wk(const float* __restrict__ qrel, const bf16* __restrict__ kh,
                     float* __restrict__ Wk) {
  int id = blockIdx.x * blockDim.x + threadIdx.x;
  if (id >= kNB * kL) return;
  int nb = id / kL, l = id % kL;
  const bf16* kr  = kh + ((long long)nb * kL + l) * kHD;
  const float* qr = qrel + nb * kHD;
  float s = 0.0f;
  for (int d = 0; d < kHD; ++d) s += qr[d] * (float)kr[d];
  Wk[id] = s;
}

// row softmax of lag-indexed, block-causal-masked logits -> bf16 weights
__global__ void k_softmax(const float* __restrict__ Wk, bf16* __restrict__ out) {
  int i  = blockIdx.x;
  int nb = blockIdx.y;
  int j  = threadIdx.x;                         // 256 threads
  const float* wrow = Wk + (long long)nb * kL;
  int ti = i >> 5, tj = j >> 5;                 // S = L/T = 32
  int lag = (i > j) ? (i - j) : (j - i);
  float v = (tj > ti) ? -INFINITY : wrow[lag];
  __shared__ float sd[256];
  sd[j] = v; __syncthreads();
  for (int s = 128; s > 0; s >>= 1) { if (j < s) sd[j] = fmaxf(sd[j], sd[j + s]); __syncthreads(); }
  float mx = sd[0]; __syncthreads();
  float e = expf(v - mx);
  sd[j] = e; __syncthreads();
  for (int s = 128; s > 0; s >>= 1) { if (j < s) sd[j] += sd[j + s]; __syncthreads(); }
  out[((long long)nb * kL + i) * kL + j] = (bf16)(e / sd[0]);
}

// merge heads: obf[n*L+l][hh*HD+d] = o[nb*L+l][d]   (f32 -> bf16)
__global__ void k_merge_heads(const float* __restrict__ o, bf16* __restrict__ obf) {
  long long id = (long long)blockIdx.x * blockDim.x + threadIdx.x;
  if (id >= (long long)kNB * kL * kHD) return;
  int d = (int)(id % kHD);
  long long r = id / kHD;
  int l  = (int)(r % kL);
  int nb = (int)(r / kL);
  int n = nb / kNH, hh = nb % kNH;
  obf[((long long)n * kL + l) * kC + hh * kHD + d] = (bf16)o[id];
}

// r1 = tok + a  (f32 out + bf16 out)
__global__ void k_residual(const float* __restrict__ tok, const float* __restrict__ a,
                           float* __restrict__ r1, bf16* __restrict__ r1b) {
  long long id = (long long)blockIdx.x * blockDim.x + threadIdx.x;
  if (id >= kMTOK * kC) return;
  float v = tok[id] + a[id];
  r1[id] = v;
  r1b[id] = (bf16)v;
}

// depthwise 3x3 SAME over (T,S) grid, S=32
__global__ void k_dwconv(const float* __restrict__ x, const float* __restrict__ dww,
                         const float* __restrict__ dwb, float* __restrict__ y) {
  long long id = (long long)blockIdx.x * blockDim.x + threadIdx.x;
  if (id >= kMTOK * kC) return;
  int c = (int)(id % kC);
  long long rr = id / kC;
  int l = (int)(rr % kL);
  int n = (int)(rr / kL);
  int t = l >> 5, s = l & 31;
  float acc = dwb[c];
  for (int dt = -1; dt <= 1; ++dt) {
    int tt = t + dt;
    if ((unsigned)tt >= (unsigned)kT) continue;
    for (int ds = -1; ds <= 1; ++ds) {
      int ss = s + ds;
      if ((unsigned)ss >= 32u) continue;
      acc += x[(((long long)n * kL) + (tt << 5) + ss) * kC + c] *
             dww[((dt + 1) * 3 + (ds + 1)) * kC + c];
    }
  }
  y[id] = acc;
}

// m = r1 + gelu(dw) * sigmoid(gate)  (in-place on dw)
__global__ void k_mixer(const float* __restrict__ r1, const float* __restrict__ gate,
                        float* __restrict__ dw) {
  long long id = (long long)blockIdx.x * blockDim.x + threadIdx.x;
  if (id >= kMTOK * kC) return;
  dw[id] = r1[id] + gelu_exact(dw[id]) * sigmoidf_(gate[id]);
}

// layernorm over C; optional fused residual (x + res)
__global__ void k_layernorm(const float* __restrict__ x, const float* __restrict__ res,
                            const float* __restrict__ w, const float* __restrict__ b,
                            float* __restrict__ out, int C) {
  long long row = blockIdx.x;
  const float* xr = x + row * C;
  const float* rr = res ? res + row * C : nullptr;
  __shared__ float s1[256], s2[256];
  float sum = 0.0f, sq = 0.0f;
  for (int i = threadIdx.x; i < C; i += 256) {
    float v = xr[i] + (rr ? rr[i] : 0.0f);
    sum += v; sq += v * v;
  }
  s1[threadIdx.x] = sum; s2[threadIdx.x] = sq;
  __syncthreads();
  for (int s = 128; s > 0; s >>= 1) {
    if (threadIdx.x < s) { s1[threadIdx.x] += s1[threadIdx.x + s]; s2[threadIdx.x] += s2[threadIdx.x + s]; }
    __syncthreads();
  }
  float mean = s1[0] / (float)C;
  float var  = s2[0] / (float)C - mean * mean;
  float inv  = rsqrtf(var + 1e-5f);
  for (int i = threadIdx.x; i < C; i += 256) {
    float v = xr[i] + (rr ? rr[i] : 0.0f);
    out[row * C + i] = (v - mean) * inv * w[i] + b[i];
  }
}

// =====================================================================
extern "C" void kernel_launch(void* const* d_in, const int* in_sizes, int n_in,
                              void* d_out, int out_size, void* d_ws, size_t ws_size,
                              hipStream_t stream) {
  (void)in_sizes; (void)n_in; (void)out_size; (void)ws_size;
  // ---- inputs in setup_inputs() insertion order ----
  const float* x = (const float*)d_in[0];
  struct AttnP { const float *in_w, *in_b, *out_w, *out_b, *conv_w, *conv_b, *qln_w, *qln_b; };
  struct MixP  { const float *dw_w, *dw_b, *gate_w, *gate_b; };
  auto F = [&](int i) { return (const float*)d_in[i]; };
  AttnP atw { F(1), F(2), F(3), F(4), F(5), F(6), F(7), F(8) };
  AttnP ath { F(9), F(10), F(11), F(12), F(13), F(14), F(15), F(16) };
  MixP  mtw { F(17), F(18), F(19), F(20) };
  MixP  mth { F(21), F(22), F(23), F(24) };
  const float *norm_tw_w = F(25), *norm_tw_b = F(26);
  const float *norm_th_w = F(27), *norm_th_b = F(28);
  const float *ffn_w1 = F(29), *ffn_b1 = F(30), *ffn_w2 = F(31), *ffn_b2 = F(32);
  const float *norm_ffn_w = F(33), *norm_ffn_b = F(34);

  // ---- workspace bump allocator (deterministic) ----
  char* ws = (char*)d_ws;
  size_t off = 0;
  auto alloc = [&](size_t bytes) { size_t o = off; off += (bytes + 255) & ~(size_t)255; return o; };
  const size_t oWinTW  = alloc((size_t)kQKV * kC * 2);
  const size_t oWinTH  = alloc((size_t)kQKV * kC * 2);
  const size_t oWoutTW = alloc((size_t)kC * kC * 2);
  const size_t oWoutTH = alloc((size_t)kC * kC * 2);
  const size_t oWcvTW  = alloc((size_t)kHD * kHDP * 2);
  const size_t oWcvTH  = alloc((size_t)kHD * kHDP * 2);
  const size_t oWgtTW  = alloc((size_t)kC * kC * 2);
  const size_t oWgtTH  = alloc((size_t)kC * kC * 2);
  const size_t oWffn1  = alloc((size_t)kDFF * kC * 2);
  const size_t oWffn2  = alloc((size_t)kC * kDFF * 2);
  const size_t oBIG    = alloc((size_t)kMTOK * kDFF * 4);  // qkv f32 | softmax bf16 | h1 f32
  const size_t oHEADS  = alloc((size_t)kMTOK * kDFF * 2);  // qh+kh+vT | h1 bf16
  const size_t oTOKF   = alloc((size_t)kMTOK * kC * 4);
  const size_t oTOKB   = alloc((size_t)kMTOK * kC * 2);
  const size_t oQT     = alloc((size_t)kNB * kL * kHD * 4); // qt | ffn out
  const size_t oQREL   = alloc((size_t)kNB * kHD * 4);
  const size_t oWK     = alloc((size_t)kNB * kL * 4);
  const size_t oO      = alloc((size_t)kNB * kL * kHD * 4);
  const size_t oOBF    = alloc((size_t)kMTOK * kC * 2);
  const size_t oA      = alloc((size_t)kMTOK * kC * 4);
  const size_t oR1     = alloc((size_t)kMTOK * kC * 4);
  const size_t oR1B    = alloc((size_t)kMTOK * kC * 2);
  const size_t oDW     = alloc((size_t)kMTOK * kC * 4);
  const size_t oGATE   = alloc((size_t)kMTOK * kC * 4);
  const size_t oN      = alloc((size_t)kMTOK * kC * 4);
  const size_t oQH = oHEADS;
  const size_t oKH = oQH + (size_t)kNB * kL * kHDP * 2;
  const size_t oVT = oKH + (size_t)kNB * kL * kHD * 2;

  const long long NTOKC = kMTOK * (long long)kC;
  auto blocks = [](long long n) { return (unsigned)((n + 255) / 256); };

  // ---- weight conversion (f32 -> bf16, every call; no cached state) ----
  auto cvt = [&](const float* src, size_t dst, long long n) {
    k_cvt_bf16<<<blocks(n), 256, 0, stream>>>(src, (bf16*)(ws + dst), n);
  };
  cvt(atw.in_w, oWinTW, (long long)kQKV * kC);
  cvt(ath.in_w, oWinTH, (long long)kQKV * kC);
  cvt(atw.out_w, oWoutTW, (long long)kC * kC);
  cvt(ath.out_w, oWoutTH, (long long)kC * kC);
  cvt(mtw.gate_w, oWgtTW, (long long)kC * kC);
  cvt(mth.gate_w, oWgtTH, (long long)kC * kC);
  cvt(ffn_w1, oWffn1, (long long)kDFF * kC);
  cvt(ffn_w2, oWffn2, (long long)kC * kDFF);
  k_cvt_pad<<<blocks((long long)kHD * kHDP), 256, 0, stream>>>(atw.conv_w, (bf16*)(ws + oWcvTW), kHD, kHD, kHDP);
  k_cvt_pad<<<blocks((long long)kHD * kHDP), 256, 0, stream>>>(ath.conv_w, (bf16*)(ws + oWcvTH), kHD, kHD, kHDP);

  // ---- one axial pass (tokens in oTOKF, layernormed result in oN) ----
  auto run_path = [&](const AttnP& ap, const MixP& mp, const float* nw, const float* nbb,
                      size_t oWin, size_t oWout, size_t oWconv, size_t oWgate) {
    float* tokF = (float*)(ws + oTOKF);
    bf16*  tokB = (bf16*)(ws + oTOKB);
    k_cvt_bf16<<<blocks(NTOKC), 256, 0, stream>>>(tokF, tokB, NTOKC);
    // qkv = tok @ in_w^T + in_b   (TDM double-buffered)
    k_gemm_tdm<0><<<dim3(kQKV / 64, (unsigned)(kMTOK / 64)), 128, 0, stream>>>(
        tokB, (const bf16*)(ws + oWin), ap.in_b, (float*)(ws + oBIG),
        (int)kMTOK, kQKV, kC);
    k_split_heads<<<blocks((long long)kNB * kL * kHDP), 256, 0, stream>>>(
        (const float*)(ws + oBIG), (bf16*)(ws + oQH), (bf16*)(ws + oKH), (bf16*)(ws + oVT));
    // qt = qh @ conv_w^T + conv_b  (K padded 48->64, N=48 -> direct kernel)
    k_gemm_bf16<1, 0><<<dim3(kHD / 16, (unsigned)((kNB * kL) / 64), 1), 128, 0, stream>>>(
        (const bf16*)(ws + oQH), (const bf16*)(ws + oWconv), ap.conv_b, (float*)(ws + oQT),
        kNB * kL, kHD, kHDP, 0, 0, 0);
    k_qt_post<<<(unsigned)(kNB * kL), 64, 0, stream>>>((float*)(ws + oQT), ap.qln_w, ap.qln_b);
    k_qrel<<<blocks((long long)kNB * kHD), 256, 0, stream>>>((const float*)(ws + oQT), (float*)(ws + oQREL));
    k_wk<<<blocks((long long)kNB * kL), 256, 0, stream>>>(
        (const float*)(ws + oQREL), (const bf16*)(ws + oKH), (float*)(ws + oWK));
    k_softmax<<<dim3(kL, kNB), 256, 0, stream>>>((const float*)(ws + oWK), (bf16*)(ws + oBIG));
    // o = softmax_w @ v   (batched, v stored transposed, N=48 -> direct kernel)
    k_gemm_bf16<1, 0><<<dim3(kHD / 16, kL / 64, kNB), 128, 0, stream>>>(
        (const bf16*)(ws + oBIG), (const bf16*)(ws + oVT), nullptr, (float*)(ws + oO),
        kL, kHD, kL, (long long)kL * kL, (long long)kHD * kL, (long long)kL * kHD);
    k_merge_heads<<<blocks((long long)kNB * kL * kHD), 256, 0, stream>>>(
        (const float*)(ws + oO), (bf16*)(ws + oOBF));
    // a = o @ out_w^T + out_b
    k_gemm_tdm<0><<<dim3(kC / 64, (unsigned)(kMTOK / 64)), 128, 0, stream>>>(
        (const bf16*)(ws + oOBF), (const bf16*)(ws + oWout), ap.out_b, (float*)(ws + oA),
        (int)kMTOK, kC, kC);
    // mixer on r1 = tok + a
    k_residual<<<blocks(NTOKC), 256, 0, stream>>>(
        tokF, (const float*)(ws + oA), (float*)(ws + oR1), (bf16*)(ws + oR1B));
    k_dwconv<<<blocks(NTOKC), 256, 0, stream>>>(
        (const float*)(ws + oR1), mp.dw_w, mp.dw_b, (float*)(ws + oDW));
    k_gemm_tdm<0><<<dim3(kC / 64, (unsigned)(kMTOK / 64)), 128, 0, stream>>>(
        (const bf16*)(ws + oR1B), (const bf16*)(ws + oWgate), mp.gate_b, (float*)(ws + oGATE),
        (int)kMTOK, kC, kC);
    k_mixer<<<blocks(NTOKC), 256, 0, stream>>>(
        (const float*)(ws + oR1), (const float*)(ws + oGATE), (float*)(ws + oDW));
    k_layernorm<<<(unsigned)kMTOK, 256, 0, stream>>>(
        (const float*)(ws + oDW), nullptr, nw, nbb, (float*)(ws + oN), kC);
  };

  // ---- pass 1: time-width axis ----
  k_make_tw<<<blocks(NTOKC), 256, 0, stream>>>(x, (float*)(ws + oTOKF));
  run_path(atw, mtw, norm_tw_w, norm_tw_b, oWinTW, oWoutTW, oWcvTW, oWgtTW);
  // ---- pass 2: time-height axis ----
  k_make_th<<<blocks(NTOKC), 256, 0, stream>>>((const float*)(ws + oN), (float*)(ws + oTOKF));
  run_path(ath, mth, norm_th_w, norm_th_b, oWinTH, oWoutTH, oWcvTH, oWgtTH);
  // ---- FFN + final layernorm ----
  k_make_final<<<blocks(NTOKC), 256, 0, stream>>>((const float*)(ws + oN), (float*)(ws + oTOKF));
  k_cvt_bf16<<<blocks(NTOKC), 256, 0, stream>>>((const float*)(ws + oTOKF), (bf16*)(ws + oTOKB), NTOKC);
  k_gemm_tdm<1><<<dim3(kDFF / 64, (unsigned)(kMTOK / 64)), 128, 0, stream>>>(
      (const bf16*)(ws + oTOKB), (const bf16*)(ws + oWffn1), ffn_b1, (float*)(ws + oBIG),
      (int)kMTOK, kDFF, kC);
  k_cvt_bf16<<<blocks(kMTOK * (long long)kDFF), 256, 0, stream>>>(
      (const float*)(ws + oBIG), (bf16*)(ws + oHEADS), kMTOK * (long long)kDFF);
  k_gemm_tdm<0><<<dim3(kC / 64, (unsigned)(kMTOK / 64)), 128, 0, stream>>>(
      (const bf16*)(ws + oHEADS), (const bf16*)(ws + oWffn2), ffn_b2, (float*)(ws + oQT),
      (int)kMTOK, kC, kDFF);
  k_layernorm<<<(unsigned)kMTOK, 256, 0, stream>>>(
      (const float*)(ws + oQT), (const float*)(ws + oTOKF),
      norm_ffn_w, norm_ffn_b, (float*)d_out, kC);
}
